// SO2_Linear_9887014716165
// MI455X (gfx1250) — matrix operand order repaired and checked by reading, not compiled, not run on hardware
//
#include <hip/hip_runtime.h>
#include <cmath>

typedef __attribute__((ext_vector_type(16))) _Float16 v16h;
typedef __attribute__((ext_vector_type(8)))  float    v8f;

#define NPTS    16384
#define DIMX    3200
#define XSTR    3208   // f16 stride: 1604 dwords, %64 == 4 -> conflict-free row access
#define OSTR    3208
#define NWTOT   1392640

// ---------------- workspace layout (bytes) ----------------
// [0        ) int   Pinv[3200]          (12.8 KB)
// [16384    ) float rot-constants[2448] (projector C/S matrices, both axes)
// [32768    ) f16   Wh[NWTOT]           (2.785 MB)
// [2818048  ) f16   Wl[NWTOT]           (2.785 MB)
// [5603328  ) float rot[l][pt][d*d]     (10.75 MB)  total ~16.4 MB
#define PINV_OFF   0
#define CONST_OFF  16384
#define WH_OFF     32768
#define WL_OFF     2818048
#define ROT_OFF    5603328

// per-l float offsets inside the constant blob: sizes 2*(2l+1)*d^2
// l=1:54  l=2:250  l=3:686  l=4:1458  -> bases 0,54,304,990
__device__ __constant__ int d_clbase[5] = {0, 0, 54, 304, 990};

// rot buffer float bases (per l), strides d*d per point
#define RB1 0
#define RB2 (9*NPTS)
#define RB3 (34*NPTS)
#define RB4 (83*NPTS)

// =====================================================================
// K0a: rotation constants.  G = Re(Q^H X Q) exactly as the reference,
// then spectral projectors so expm(theta*G) = sum_m cos(m t)C_m + sin(m t)S_m.
// =====================================================================
__global__ __launch_bounds__(256) void setup_consts(float* cst) {
  __shared__ double Xr[4][81], Xi[4][81], Qr[4][81], Qi[4][81];
  __shared__ double Tr[4][81], Ti[4][81], G[4][81], M2[4][81];
  __shared__ double P[4][5][81], Pn[4][5][81];

  const int tid = threadIdx.x;
  const int li  = tid >> 6;          // 0..3 -> l = li+1
  const int g   = tid & 63;
  const int l   = li + 1;
  const int D   = 2*l + 1;
  const int DD  = D*D;

  for (int axis = 0; axis < 2; ++axis) {   // axis0: G[1] (alpha), axis1: G[0] (beta)
    for (int e = g; e < DD; e += 64) { Xr[li][e]=0; Xi[li][e]=0; Qr[li][e]=0; Qi[li][e]=0; }
    __syncthreads();
    if (g == 0) {
      if (axis == 0) {                       // X1 = diag(i*m), m=-l..l
        for (int a = 0; a < D; ++a) Xi[li][a*D+a] = (double)(a - l);
      } else {                               // X0 = 0.5*(raising+lowering)
        for (int a = 0; a < D-1; ++a) {
          double m1 = (double)(a - l);
          Xr[li][(a+1)*D + a] += -0.5 * sqrt((double)l*(l+1) - m1*(m1+1.0));
          double m2 = (double)(a + 1 - l);
          Xr[li][a*D + (a+1)] +=  0.5 * sqrt((double)l*(l+1) - m2*(m2-1.0));
        }
      }
      const double is2 = 0.70710678118654752440;
      for (int m = -l; m < 0; ++m) { int r = l+m; Qr[li][r*D+(l-m)] = is2; Qi[li][r*D+(l+m)] = -is2; }
      Qr[li][l*D+l] = 1.0;
      for (int m = 1; m <= l; ++m) { int r = l+m; double sg = (m & 1) ? -1.0 : 1.0;
        Qr[li][r*D+(l+m)] = sg*is2; Qi[li][r*D+(l-m)] = sg*is2; }
    }
    __syncthreads();
    // T = X*Q (complex)
    for (int e = g; e < DD; e += 64) {
      int i = e / D, j = e % D; double sr = 0, si = 0;
      for (int k = 0; k < D; ++k) {
        double xr = Xr[li][i*D+k], xi = Xi[li][i*D+k];
        double qr = Qr[li][k*D+j], qi = Qi[li][k*D+j];
        sr += xr*qr - xi*qi; si += xr*qi + xi*qr;
      }
      Tr[li][e] = sr; Ti[li][e] = si;
    }
    __syncthreads();
    // G = Re(Q^H T)
    for (int e = g; e < DD; e += 64) {
      int a = e / D, b = e % D; double s = 0;
      for (int p = 0; p < D; ++p) s += Qr[li][p*D+a]*Tr[li][p*D+b] + Qi[li][p*D+a]*Ti[li][p*D+b];
      G[li][e] = s;
    }
    __syncthreads();
    // M2 = G*G
    for (int e = g; e < DD; e += 64) {
      int i = e / D, j = e % D; double s = 0;
      for (int k = 0; k < D; ++k) s += G[li][i*D+k]*G[li][k*D+j];
      M2[li][e] = s;
    }
    __syncthreads();
    for (int me = 0; me <= l; ++me)
      for (int e = g; e < DD; e += 64) P[li][me][e] = (e/D == e%D) ? 1.0 : 0.0;
    __syncthreads();
    for (int step = 0; step <= 4; ++step) {   // uniform trip count for barriers
      for (int me = 0; me <= l; ++me) {
        for (int e = g; e < DD; e += 64) {
          double v;
          if (step > l || me == step) v = P[li][me][e];
          else {
            int i = e / D, j = e % D; double s = 0;
            for (int k = 0; k < D; ++k) {
              double mm = M2[li][k*D+j] + ((k == j) ? (double)(step*step) : 0.0);
              s += P[li][me][i*D+k] * mm;
            }
            v = s / ((double)(step*step) - (double)(me*me));
          }
          Pn[li][me][e] = v;
        }
      }
      __syncthreads();
      for (int me = 0; me <= l; ++me)
        for (int e = g; e < DD; e += 64) P[li][me][e] = Pn[li][me][e];
      __syncthreads();
    }
    float* dst = cst + d_clbase[l] + axis * (2*l+1) * DD;
    for (int me = 0; me <= l; ++me)
      for (int e = g; e < DD; e += 64) dst[me*DD + e] = (float)P[li][me][e];
    for (int me = 1; me <= l; ++me)
      for (int e = g; e < DD; e += 64) {
        int i = e / D, j = e % D; double s = 0;
        for (int k = 0; k < D; ++k) s += G[li][i*D+k] * P[li][me][k*D+j];
        dst[(l+me)*DD + e] = (float)(s / (double)me);
      }
    __syncthreads();
  }
}

// =====================================================================
// K0b: M_IDX permutation (closed form) + weight split to f16 hi/lo.
// =====================================================================
__global__ __launch_bounds__(256) void setup_tables(
    const float* __restrict__ w0, const float* __restrict__ w1,
    const float* __restrict__ w2, const float* __restrict__ w3,
    const float* __restrict__ w4,
    int* __restrict__ pinv, _Float16* __restrict__ wh, _Float16* __restrict__ wl) {
  for (int i = blockIdx.x*blockDim.x + threadIdx.x; i < NWTOT; i += gridDim.x*blockDim.x) {
    float w;
    if      (i <  409600) w = w0[i];
    else if (i <  933888) w = w1[i -  409600];
    else if (i < 1228800) w = w2[i -  933888];
    else if (i < 1359872) w = w3[i - 1228800];
    else                  w = w4[i - 1359872];
    _Float16 h = (_Float16)w;
    wh[i] = h;
    wl[i] = (_Float16)(w - (float)h);
  }
  int i = blockIdx.x*blockDim.x + threadIdx.x;
  if (i < DIMX) {
    int l, off;
    if      (i <  128) { l = 0; off = 0;    }
    else if (i <  512) { l = 1; off = 128;  }
    else if (i < 1152) { l = 2; off = 512;  }
    else if (i < 2048) { l = 3; off = 1152; }
    else               { l = 4; off = 2048; }
    const int d  = 2*l + 1;
    const int r  = i - off;
    const int ch = r / d;
    const int ii = r % d;
    const int m  = (ii > l) ? (ii - l) : (l - ii);
    const int baseм[5] = {0, 640, 1664, 2432, 2944};
    const int before = (m == 0) ? l*128 : (l - m)*256;
    const int within = (m == 0) ? ch : 2*ch + ((ii > l) ? 1 : 0);
    pinv[i] = baseм[m] + before + within;
  }
}

// =====================================================================
// K1: per-point Wigner rotations rot[l] = expm(a*G1) @ expm(b*G0)
// =====================================================================
template<int L>
__device__ __forceinline__ void rot_point(const float* __restrict__ cstl,
                                          const float* ca, const float* sa,
                                          const float* cb, const float* sb,
                                          float* __restrict__ outp) {
  constexpr int D = 2*L + 1;
  float Ra[D*D], Rb[D*D];
  const float* A0 = cstl;                    // alpha axis
  const float* A1 = cstl + (2*L+1)*D*D;      // beta axis
#pragma unroll
  for (int e = 0; e < D*D; ++e) {
    float va = ca[0]*A0[e];
    float vb = cb[0]*A1[e];
#pragma unroll
    for (int m = 1; m <= L; ++m) {
      va += ca[m]*A0[m*D*D + e] + sa[m]*A0[(L+m)*D*D + e];
      vb += cb[m]*A1[m*D*D + e] + sb[m]*A1[(L+m)*D*D + e];
    }
    Ra[e] = va; Rb[e] = vb;
  }
#pragma unroll
  for (int i = 0; i < D; ++i)
#pragma unroll
    for (int j = 0; j < D; ++j) {
      float s = 0.f;
#pragma unroll
      for (int k = 0; k < D; ++k) s += Ra[i*D+k]*Rb[k*D+j];
      outp[i*D+j] = s;
    }
}

__global__ __launch_bounds__(256) void rot_kernel(const float* __restrict__ R,
                                                  const float* __restrict__ cst,
                                                  float* __restrict__ rotb) {
  int p = blockIdx.x*blockDim.x + threadIdx.x;
  if (p >= NPTS) return;
  float r0 = R[3*p], r1 = R[3*p+1], r2 = R[3*p+2];
  float nrm = fmaxf(sqrtf(r0*r0 + r1*r1 + r2*r2), 1e-12f);
  // v = R[:, [1,2,0]]: v0=R1, v1=R2, v2=R0
  float v0 = fminf(fmaxf(r1/nrm, -1.f), 1.f);
  float v1 = fminf(fmaxf(r2/nrm, -1.f), 1.f);
  float v2 = fminf(fmaxf(r0/nrm, -1.f), 1.f);
  float beta  = acosf(v1);
  float alpha = atan2f(v0, v2);
  float ca[5], sa[5], cb[5], sb[5];
  ca[0] = 1.f; sa[0] = 0.f; cb[0] = 1.f; sb[0] = 0.f;
  ca[1] = cosf(alpha); sa[1] = sinf(alpha);
  cb[1] = cosf(beta);  sb[1] = sinf(beta);
#pragma unroll
  for (int m = 2; m <= 4; ++m) {      // Chebyshev recurrences
    ca[m] = 2.f*ca[1]*ca[m-1] - ca[m-2];
    sa[m] = 2.f*ca[1]*sa[m-1] - sa[m-2];
    cb[m] = 2.f*cb[1]*cb[m-1] - cb[m-2];
    sb[m] = 2.f*cb[1]*sb[m-1] - sb[m-2];
  }
  rot_point<1>(cst +   0, ca, sa, cb, sb, rotb + RB1 + (size_t)p*9);
  rot_point<2>(cst +  54, ca, sa, cb, sb, rotb + RB2 + (size_t)p*25);
  rot_point<3>(cst + 304, ca, sa, cb, sb, rotb + RB3 + (size_t)p*49);
  rot_point<4>(cst + 990, ca, sa, cb, sb, rotb + RB4 + (size_t)p*81);
}

// =====================================================================
// K2: fused main kernel. 16 points/block, 8 waves, LDS-resident panel.
// =====================================================================
static __device__ __forceinline__ v8f wmma_f16(v16h a, v16h b, v8f c) {
  return __builtin_amdgcn_wmma_f32_16x16x32_f16(false, a, false, b, (short)0, c, false, false);
}

// A tile (16x32 f16): lane = lrow + 16*lhi; row M=lrow; K = lhi*8 + {0..7, 16..23}
static __device__ __forceinline__ v16h loadA(const _Float16* xp, int lrow, int lhi, int gcol) {
  const _Float16* p = xp + lrow*XSTR + gcol + lhi*8;
  union { v16h v; uint4 q[2]; } u;
  u.q[0] = *(const uint4*)(p);
  u.q[1] = *(const uint4*)(p + 16);
  return u.v;
}
// B tile (32x16 f16): lane col N=lrow; K = lhi*8 + {0..7, 16..23}
static __device__ __forceinline__ v16h loadB(const _Float16* __restrict__ W, int kdim,
                                             int otile, int kk, int lrow, int lhi) {
  const _Float16* p = W + (size_t)(otile*16 + lrow)*kdim + kk*32 + lhi*8;
  union { v16h v; uint4 q[2]; } u;
  u.q[0] = *(const uint4*)(p);
  u.q[1] = *(const uint4*)(p + 16);
  return u.v;
}
// C/D tile: VGPR j -> row j + 8*lhi, col lrow
static __device__ __forceinline__ void storeTile(float* outp, int gtile, int lrow, int lhi, v8f v) {
  int col = gtile*16 + lrow;
#pragma unroll
  for (int j = 0; j < 8; ++j) outp[(lhi*8 + j)*OSTR + col] = v[j];
}

template<int L>
__device__ __forceinline__ void fwd_rot(int ch, const float* __restrict__ xrow,
                                        const float* rl, const int* __restrict__ pinv,
                                        _Float16* xr) {
  constexpr int D   = 2*L + 1;
  constexpr int OFF = (L==1) ? 128 : (L==2) ? 512 : (L==3) ? 1152 : 2048;
  float xv[D];
#pragma unroll
  for (int j = 0; j < D; ++j) xv[j] = xrow[OFF + ch*D + j];
#pragma unroll
  for (int i = 0; i < D; ++i) {          // x_ = x_row @ rot  (einsum nji,nmj->nmi)
    float s = 0.f;
#pragma unroll
    for (int j = 0; j < D; ++j) s += xv[j] * rl[j*D + i];
    xr[pinv[OFF + ch*D + i]] = (_Float16)s;
  }
}

template<int L>
__device__ __forceinline__ void bwd_rot(int ch, const float* rl, const int* __restrict__ pinv,
                                        const float* opr, float* __restrict__ orow) {
  constexpr int D   = 2*L + 1;
  constexpr int OFF = (L==1) ? 128 : (L==2) ? 512 : (L==3) ? 1152 : 2048;
  float t[D];
#pragma unroll
  for (int j = 0; j < D; ++j) t[j] = opr[pinv[OFF + ch*D + j]];
#pragma unroll
  for (int i = 0; i < D; ++i) {          // out = rot @ out_pre_row (einsum nij,nmj->nmi)
    float s = 0.f;
#pragma unroll
    for (int j = 0; j < D; ++j) s += rl[i*D + j] * t[j];
    orow[OFF + ch*D + i] = s;
  }
}

__global__ __launch_bounds__(256) void main_kernel(
    const float* __restrict__ x, const float* __restrict__ bm0,
    const _Float16* __restrict__ Wh, const _Float16* __restrict__ Wl,
    const int* __restrict__ pinv,
    const float* __restrict__ rot1, const float* __restrict__ rot2,
    const float* __restrict__ rot3, const float* __restrict__ rot4,
    float* __restrict__ out) {
  __shared__ __align__(16) _Float16 xperm[16*XSTR];   // 102.7 KB
  __shared__ __align__(16) float    outpre[16*OSTR];  // 205.3 KB
  __shared__ float rotl[2624];                        //  10.5 KB  (total 318.5 KB / 320 KB WGP)

  const int p0   = blockIdx.x * 16;
  const int tid  = threadIdx.x;
  const int wave = tid >> 5;
  const int lane = tid & 31;
  const int lrow = lane & 15;
  const int lhi  = lane >> 4;

  // ---- phase 0: stage per-point rot matrices into LDS ----
  for (int idx = tid; idx < 2624; idx += 256) {
    int base, dd; const float* src;
    if      (idx <  144) { base = 0;    dd =  9; src = rot1; }
    else if (idx <  544) { base = 144;  dd = 25; src = rot2; }
    else if (idx < 1328) { base = 544;  dd = 49; src = rot3; }
    else                 { base = 1328; dd = 81; src = rot4; }
    int r = idx - base;
    rotl[idx] = src[(size_t)(p0 + r/dd)*dd + (r % dd)];
  }
  __syncthreads();

  // ---- phase 1: rotate x forward, write permuted f16 panel ----
  for (int task = tid; task < 2048; task += 256) {
    int pt = task >> 7, ch = task & 127;
    const float* xrow = x + (size_t)(p0 + pt)*DIMX;
    _Float16* xr = xperm + pt*XSTR;
    xr[ch] = (_Float16)0.f;                      // l=0 block of x_ is zero; Pinv[ch]==ch
    fwd_rot<1>(ch, xrow, rotl +    0 + pt*9,  pinv, xr);
    fwd_rot<2>(ch, xrow, rotl +  144 + pt*25, pinv, xr);
    fwd_rot<3>(ch, xrow, rotl +  544 + pt*49, pinv, xr);
    fwd_rot<4>(ch, xrow, rotl + 1328 + pt*81, pinv, xr);
  }
  __syncthreads();

  // ---- phase 2: WMMA GEMMs (120 work units round-robined over 8 waves) ----
  for (int u = wave; u < 120; u += 8) {
    if (u < 40) {                                // m=0: out = Xg0 @ w0^T + b
      v8f acc = {};
      for (int kk = 0; kk < 20; ++kk) {          // K = 640
        v16h a  = loadA(xperm, lrow, lhi, kk*32);
        v16h bh = loadB(Wh, 640, u, kk, lrow, lhi);
        v16h bl = loadB(Wl, 640, u, kk, lrow, lhi);
        acc = wmma_f16(a, bh, acc);
        acc = wmma_f16(a, bl, acc);
      }
      float bias = bm0[u*16 + lrow];
#pragma unroll
      for (int j = 0; j < 8; ++j) acc[j] += bias;
      storeTile(outpre, u, lrow, lhi, acc);
    } else {                                     // m>=1: paired complex combine
      int u2 = u - 40, t, base, k, woff;
      if      (u2 < 32) { t = u2;      base =  640; k = 512; woff =  409600; }
      else if (u2 < 56) { t = u2 - 32; base = 1664; k = 384; woff =  933888; }
      else if (u2 < 72) { t = u2 - 56; base = 2432; k = 256; woff = 1228800; }
      else              { t = u2 - 72; base = 2944; k = 128; woff = 1359872; }
      const int kt = k >> 5;
      v8f c0l = {}, c0h = {}, c1l = {}, c1h = {};
      for (int kk = 0; kk < kt; ++kk) {
        v16h a0  = loadA(xperm, lrow, lhi, base + kk*32);          // c = 0
        v16h a1  = loadA(xperm, lrow, lhi, base + k + kk*32);      // c = 1
        v16h blh = loadB(Wh + woff, k, t,            kk, lrow, lhi);
        v16h bll = loadB(Wl + woff, k, t,            kk, lrow, lhi);
        v16h bhh = loadB(Wh + woff, k, t + (k>>4),   kk, lrow, lhi);
        v16h bhl = loadB(Wl + woff, k, t + (k>>4),   kk, lrow, lhi);
        c0l = wmma_f16(a0, blh, c0l); c0l = wmma_f16(a0, bll, c0l);
        c0h = wmma_f16(a0, bhh, c0h); c0h = wmma_f16(a0, bhl, c0h);
        c1l = wmma_f16(a1, blh, c1l); c1l = wmma_f16(a1, bll, c1l);
        c1h = wmma_f16(a1, bhh, c1h); c1h = wmma_f16(a1, bhl, c1h);
      }
      v8f yr = c0l - c1h;                        // y_r = Y0[:, :k] - Y1[:, k:]
      v8f yi = c0h + c1l;                        // y_i = Y0[:, k:] + Y1[:, :k]
      storeTile(outpre, (base >> 4) + t,            lrow, lhi, yr);
      storeTile(outpre, (base >> 4) + (k >> 4) + t, lrow, lhi, yi);
    }
  }
  __syncthreads();

  // ---- phase 3: rotate back and write final output ----
  for (int task = tid; task < 2048; task += 256) {
    int pt = task >> 7, ch = task & 127;
    const float* opr = outpre + pt*OSTR;
    float* orow = out + (size_t)(p0 + pt)*DIMX;
    orow[ch] = opr[ch];                          // l=0 passthrough (Pinv[ch]==ch)
    bwd_rot<1>(ch, rotl +    0 + pt*9,  pinv, opr, orow);
    bwd_rot<2>(ch, rotl +  144 + pt*25, pinv, opr, orow);
    bwd_rot<3>(ch, rotl +  544 + pt*49, pinv, opr, orow);
    bwd_rot<4>(ch, rotl + 1328 + pt*81, pinv, opr, orow);
  }
}

// =====================================================================
extern "C" void kernel_launch(void* const* d_in, const int* in_sizes, int n_in,
                              void* d_out, int out_size, void* d_ws, size_t ws_size,
                              hipStream_t stream) {
  (void)in_sizes; (void)n_in; (void)out_size; (void)ws_size;  // needs ~16.4 MB ws
  const float* x  = (const float*)d_in[0];
  const float* R  = (const float*)d_in[1];
  const float* w0 = (const float*)d_in[2];
  const float* b0 = (const float*)d_in[3];
  const float* w1 = (const float*)d_in[4];
  const float* w2 = (const float*)d_in[5];
  const float* w3 = (const float*)d_in[6];
  const float* w4 = (const float*)d_in[7];
  float* out = (float*)d_out;

  char* ws = (char*)d_ws;
  int*      pinv = (int*)     (ws + PINV_OFF);
  float*    cst  = (float*)   (ws + CONST_OFF);
  _Float16* wh   = (_Float16*)(ws + WH_OFF);
  _Float16* wl   = (_Float16*)(ws + WL_OFF);
  float*    rotb = (float*)   (ws + ROT_OFF);

  setup_consts<<<1, 256, 0, stream>>>(cst);
  setup_tables<<<5440, 256, 0, stream>>>(w0, w1, w2, w3, w4, pinv, wh, wl);
  rot_kernel<<<NPTS/256, 256, 0, stream>>>(R, cst, rotb);
  main_kernel<<<NPTS/16, 256, 0, stream>>>(x, b0, wh, wl, pinv,
                                           rotb + RB1, rotb + RB2, rotb + RB3, rotb + RB4,
                                           out);
}
